// MultiHeadAttention_90074054132478
// MI455X (gfx1250) — compile-verified, matrix operand order checked by reading, not run on hardware
//
#include <hip/hip_runtime.h>

typedef __attribute__((ext_vector_type(16))) __bf16 v16bf;
typedef __attribute__((ext_vector_type(8)))  __bf16 v8bf;
typedef __attribute__((ext_vector_type(4)))  __bf16 v4bf;
typedef __attribute__((ext_vector_type(8)))  float  v8f;

#define B_  4
#define S_  2048
#define D_  1024
#define H_  16
#define HD_ 64

// ---------------------------------------------------------------------------
// Fragment loader for 16-bit A/B WMMA operands (16x32 / 32x16 bf16).
// Per CDNA5 ISA 7.12.2: lane L (row/col = L%16, half = L/16) holds, for a
// 32-deep K chunk, elements K = half*8 + [0..7] and K = 16 + half*8 + [0..7].
// Both chunks are 16-byte contiguous loads from the row base pointer.
// ---------------------------------------------------------------------------
__device__ __forceinline__ v16bf load_frag(const __bf16* base, int half) {
    union { v16bf v; v8bf h[2]; } u;
    u.h[0] = *(const v8bf*)(base + half * 8);
    u.h[1] = *(const v8bf*)(base + 16 + half * 8);
    return u.v;
}

#define WMMA_BF16(A, B, C) \
    __builtin_amdgcn_wmma_f32_16x16x32_bf16(false, (A), false, (B), (short)0, (C), false, false)

// ---------------------------------------------------------------------------
// Stage 0: f32 -> bf16 cast (4 elements / thread)
// ---------------------------------------------------------------------------
__global__ void mha_cast_bf16_kernel(const float* __restrict__ src,
                                     __bf16* __restrict__ dst, int n4) {
    int i = blockIdx.x * blockDim.x + threadIdx.x;
    if (i < n4) {
        float4 f = ((const float4*)src)[i];
        v4bf o;
        o[0] = (__bf16)f.x; o[1] = (__bf16)f.y;
        o[2] = (__bf16)f.z; o[3] = (__bf16)f.w;
        ((v4bf*)dst)[i] = o;
    }
}

// ---------------------------------------------------------------------------
// Stage 1: QKV projection (y = x @ W^T) with fused RoPE on Q/K.
// One wave -> 32 rows x 64 cols (one full head slice, two 16-row subtiles
// sharing the 4 B-fragments: 8 WMMAs per 6 fragment loads).
// Q: [B,H,S,HD] bf16, scaled by 1/sqrt(HD), RoPE applied
// K: [B,H,S,HD] bf16, RoPE applied
// V: [B,H,HD,S] bf16 (transposed so attention B-fragments are contiguous)
// ---------------------------------------------------------------------------
__global__ __launch_bounds__(128) void mha_qkv_rope_kernel(
    const __bf16* __restrict__ xb,
    const __bf16* __restrict__ Wq, const __bf16* __restrict__ Wk,
    const __bf16* __restrict__ Wv,
    __bf16* __restrict__ Qout, __bf16* __restrict__ Kout,
    __bf16* __restrict__ Vout)
{
    const int ROWT = (B_ * S_) / 32;                 // 256 row tiles (32 rows)
    const int wave = blockIdx.x * (blockDim.x >> 5) + (threadIdx.x >> 5);
    const int lane = threadIdx.x & 31;
    const int half = lane >> 4;
    const int l16  = lane & 15;

    const int mat = wave / (H_ * ROWT);              // 0=Q 1=K 2=V
    const int rem = wave % (H_ * ROWT);
    const int h   = rem / ROWT;
    const int rt  = rem % ROWT;

    const __bf16* W = (mat == 0) ? Wq : (mat == 1) ? Wk : Wv;

    const __bf16* arow0 = xb + (size_t)(rt * 32 + l16) * D_;
    const __bf16* arow1 = xb + (size_t)(rt * 32 + 16 + l16) * D_;
    const __bf16* brow[4];
#pragma unroll
    for (int t = 0; t < 4; ++t)
        brow[t] = W + (size_t)(h * HD_ + t * 16 + l16) * D_;

    v8f acc[2][4] = {};
    for (int k0 = 0; k0 < D_; k0 += 32) {
        v16bf a0 = load_frag(arow0 + k0, half);
        v16bf a1 = load_frag(arow1 + k0, half);
#pragma unroll
        for (int t = 0; t < 4; ++t) {
            v16bf b = load_frag(brow[t] + k0, half);
            acc[0][t] = WMMA_BF16(a0, b, acc[0][t]);
            acc[1][t] = WMMA_BF16(a1, b, acc[1][t]);
        }
    }

    if (mat == 2) {
        // V: store transposed [B,H,HD,S]
#pragma unroll
        for (int sub = 0; sub < 2; ++sub)
#pragma unroll
            for (int t = 0; t < 4; ++t) {
                int hd = t * 16 + l16;
#pragma unroll
                for (int i = 0; i < 8; ++i) {
                    int r = rt * 32 + sub * 16 + i + 8 * half;   // row in [0, B*S)
                    int b = r >> 11;                             // / S_
                    int s = r & (S_ - 1);
                    Vout[(((size_t)b * H_ + h) * HD_ + hd) * S_ + s] =
                        (__bf16)acc[sub][t][i];
                }
            }
    } else {
        // RoPE: partner column hd±32 is accumulator tile t^2, same lane/element.
        __bf16* out = (mat == 0) ? Qout : Kout;
        const float scl = (mat == 0) ? 0.125f : 1.0f;    // fold 1/sqrt(64) into Q
#pragma unroll
        for (int sub = 0; sub < 2; ++sub) {
            float rot[4][8];
#pragma unroll
            for (int t = 0; t < 4; ++t) {
                int hd = t * 16 + l16;
                int j  = hd & 31;
                // inv_freq = 10000^(-j/32) = exp(-j * ln(1e4)/32)
                float inv_freq = __expf(-(float)j * (9.210340371976184f / 32.0f));
#pragma unroll
                for (int i = 0; i < 8; ++i) {
                    int   r    = rt * 32 + sub * 16 + i + 8 * half;
                    float spos = (float)(r & (S_ - 1));
                    float ang  = spos * inv_freq;
                    float c = __cosf(ang), sn = __sinf(ang);
                    float other = (t < 2) ? -acc[sub][t + 2][i] : acc[sub][t - 2][i];
                    rot[t][i] = acc[sub][t][i] * c + other * sn;
                }
            }
#pragma unroll
            for (int t = 0; t < 4; ++t) {
                int hd = t * 16 + l16;
#pragma unroll
                for (int i = 0; i < 8; ++i) {
                    int r = rt * 32 + sub * 16 + i + 8 * half;
                    int b = r >> 11;
                    int s = r & (S_ - 1);
                    out[(((size_t)b * H_ + h) * S_ + s) * HD_ + hd] =
                        (__bf16)(rot[t][i] * scl);
                }
            }
        }
    }
}

// ---------------------------------------------------------------------------
// Stage 2: Flash-style attention. One wave = one 16-query tile of one (b,h).
// Streams 32 keys per iteration: 2 score tiles (2 chained WMMAs each over
// HD=64), online softmax (cross-lane reduce inside 16-lane halves), P goes
// through per-wave LDS to convert D-layout -> A-fragment layout, then 4 WMMAs
// accumulate O (16x64 f32).
// ---------------------------------------------------------------------------
__global__ __launch_bounds__(128) void mha_attn_kernel(
    const __bf16* __restrict__ Qb, const __bf16* __restrict__ Kb,
    const __bf16* __restrict__ Vt, __bf16* __restrict__ Ob)
{
    __shared__ __align__(16) __bf16 plds[4][16 * 32];

    const int wv   = threadIdx.x >> 5;
    const int lane = threadIdx.x & 31;
    const int half = lane >> 4;
    const int l16  = lane & 15;

    const int QT   = S_ / 16;                        // 128 query tiles
    const int wave = blockIdx.x * 4 + wv;
    const int bh   = wave / QT;
    const int qt   = wave % QT;
    const int b    = bh / H_;
    const int h    = bh % H_;

    const __bf16* qbase = Qb + (((size_t)b * H_ + h) * S_) * HD_;
    const __bf16* kbase = Kb + (((size_t)b * H_ + h) * S_) * HD_;
    const __bf16* vbase = Vt + (((size_t)b * H_ + h) * HD_) * S_;

    const __bf16* qrow = qbase + (size_t)(qt * 16 + l16) * HD_;
    const v16bf qa0 = load_frag(qrow + 0, half);
    const v16bf qa1 = load_frag(qrow + 32, half);

    v8f   O[4] = {};
    float m[8], l[8];
#pragma unroll
    for (int i = 0; i < 8; ++i) { m[i] = -1e30f; l[i] = 0.0f; }

    __bf16*       pl    = plds[wv];
    const __bf16* plrow = pl + l16 * 32;

    for (int kt = 0; kt < S_ / 32; ++kt) {
        // ---- scores: 16 queries x 32 keys ----
        v8f s0 = {}, s1 = {};
        {
            const __bf16* krow0 = kbase + (size_t)(kt * 32 + l16) * HD_;
            const __bf16* krow1 = kbase + (size_t)(kt * 32 + 16 + l16) * HD_;
            v16bf kb;
            kb = load_frag(krow0 + 0, half);  s0 = WMMA_BF16(qa0, kb, s0);
            kb = load_frag(krow0 + 32, half); s0 = WMMA_BF16(qa1, kb, s0);
            kb = load_frag(krow1 + 0, half);  s1 = WMMA_BF16(qa0, kb, s1);
            kb = load_frag(krow1 + 32, half); s1 = WMMA_BF16(qa1, kb, s1);
        }

        // ---- online softmax (row i+8*half lives in element i, 16 lanes) ----
        float alpha[8];
#pragma unroll
        for (int i = 0; i < 8; ++i) {
            float v = fmaxf(s0[i], s1[i]);
            v = fmaxf(v, __shfl_xor(v, 1, 32));
            v = fmaxf(v, __shfl_xor(v, 2, 32));
            v = fmaxf(v, __shfl_xor(v, 4, 32));
            v = fmaxf(v, __shfl_xor(v, 8, 32));
            float mn = fmaxf(m[i], v);
            alpha[i] = __expf(m[i] - mn);
            m[i] = mn;

            float p0 = __expf(s0[i] - mn);
            float p1 = __expf(s1[i] - mn);
            int row = i + 8 * half;
            pl[row * 32 + l16]      = (__bf16)p0;    // D-layout -> row-major LDS
            pl[row * 32 + 16 + l16] = (__bf16)p1;

            float rs = p0 + p1;
            rs += __shfl_xor(rs, 1, 32);
            rs += __shfl_xor(rs, 2, 32);
            rs += __shfl_xor(rs, 4, 32);
            rs += __shfl_xor(rs, 8, 32);
            l[i] = l[i] * alpha[i] + rs;
        }

        // ---- rescale O, then O += P @ V ----
#pragma unroll
        for (int t = 0; t < 4; ++t)
#pragma unroll
            for (int i = 0; i < 8; ++i)
                O[t][i] *= alpha[i];

        v16bf pa = load_frag(plrow, half);           // per-wave DS ops in-order
#pragma unroll
        for (int t = 0; t < 4; ++t) {
            const __bf16* vrow = vbase + (size_t)(t * 16 + l16) * S_ + kt * 32;
            v16bf vb = load_frag(vrow, half);
            O[t] = WMMA_BF16(pa, vb, O[t]);
        }
    }

    // ---- normalize and store bf16 context [B,S,D] ----
#pragma unroll
    for (int t = 0; t < 4; ++t) {
        int hd = t * 16 + l16;
#pragma unroll
        for (int i = 0; i < 8; ++i) {
            int srow = qt * 16 + i + 8 * half;
            Ob[((size_t)(b * S_ + srow)) * D_ + h * HD_ + hd] = (__bf16)(O[t][i] / l[i]);
        }
    }
}

// ---------------------------------------------------------------------------
// Stage 3: output projection out = ctx @ Wo^T, f32 result.
// One wave -> 32 rows x 64 cols (8 WMMAs per 6 fragment loads).
// ---------------------------------------------------------------------------
__global__ __launch_bounds__(128) void mha_oproj_kernel(
    const __bf16* __restrict__ Ab, const __bf16* __restrict__ Wo,
    float* __restrict__ out)
{
    const int NT   = D_ / 64;                        // 16 column strips
    const int wave = blockIdx.x * 4 + (threadIdx.x >> 5);
    const int lane = threadIdx.x & 31;
    const int half = lane >> 4;
    const int l16  = lane & 15;
    const int rt   = wave / NT;
    const int ct   = wave % NT;

    const __bf16* arow0 = Ab + (size_t)(rt * 32 + l16) * D_;
    const __bf16* arow1 = Ab + (size_t)(rt * 32 + 16 + l16) * D_;
    const __bf16* brow[4];
#pragma unroll
    for (int t = 0; t < 4; ++t)
        brow[t] = Wo + (size_t)(ct * 64 + t * 16 + l16) * D_;

    v8f acc[2][4] = {};
    for (int k0 = 0; k0 < D_; k0 += 32) {
        v16bf a0 = load_frag(arow0 + k0, half);
        v16bf a1 = load_frag(arow1 + k0, half);
#pragma unroll
        for (int t = 0; t < 4; ++t) {
            v16bf bb = load_frag(brow[t] + k0, half);
            acc[0][t] = WMMA_BF16(a0, bb, acc[0][t]);
            acc[1][t] = WMMA_BF16(a1, bb, acc[1][t]);
        }
    }
#pragma unroll
    for (int sub = 0; sub < 2; ++sub)
#pragma unroll
        for (int t = 0; t < 4; ++t)
#pragma unroll
            for (int i = 0; i < 8; ++i) {
                int r = rt * 32 + sub * 16 + i + 8 * half;
                out[(size_t)r * D_ + ct * 64 + t * 16 + l16] = acc[sub][t][i];
            }
}

// ---------------------------------------------------------------------------
// Host launcher
// ---------------------------------------------------------------------------
extern "C" void kernel_launch(void* const* d_in, const int* in_sizes, int n_in,
                              void* d_out, int out_size, void* d_ws, size_t ws_size,
                              hipStream_t stream) {
    const float* x  = (const float*)d_in[0];
    const float* Wq = (const float*)d_in[1];
    const float* Wk = (const float*)d_in[2];
    const float* Wv = (const float*)d_in[3];
    const float* Wo = (const float*)d_in[4];

    char*  ws  = (char*)d_ws;
    size_t off = 0;
    auto alloc = [&](size_t bytes) -> char* {
        char* p = ws + off;
        off += (bytes + 255) & ~(size_t)255;
        return p;
    };

    const size_t NX = (size_t)B_ * S_ * D_;          // 8388608
    const size_t NW = (size_t)D_ * D_;               // 1048576

    __bf16* xb   = (__bf16*)alloc(NX * 2);
    __bf16* wqb  = (__bf16*)alloc(NW * 2);
    __bf16* wkb  = (__bf16*)alloc(NW * 2);
    __bf16* wvb  = (__bf16*)alloc(NW * 2);
    __bf16* wob  = (__bf16*)alloc(NW * 2);
    __bf16* qb   = (__bf16*)alloc(NX * 2);           // [B,H,S,HD]
    __bf16* kb   = (__bf16*)alloc(NX * 2);           // [B,H,S,HD]
    __bf16* vtb  = (__bf16*)alloc(NX * 2);           // [B,H,HD,S]
    __bf16* ctxb = (__bf16*)alloc(NX * 2);           // [B,S,D]

    // Stage 0: casts
    {
        int n4 = (int)(NX / 4);
        mha_cast_bf16_kernel<<<(n4 + 255) / 256, 256, 0, stream>>>(x, xb, n4);
        int w4 = (int)(NW / 4);
        mha_cast_bf16_kernel<<<(w4 + 255) / 256, 256, 0, stream>>>(Wq, wqb, w4);
        mha_cast_bf16_kernel<<<(w4 + 255) / 256, 256, 0, stream>>>(Wk, wkb, w4);
        mha_cast_bf16_kernel<<<(w4 + 255) / 256, 256, 0, stream>>>(Wv, wvb, w4);
        mha_cast_bf16_kernel<<<(w4 + 255) / 256, 256, 0, stream>>>(Wo, wob, w4);
    }

    // Stage 1: QKV + RoPE  (3 mats * 16 heads * 256 row tiles = 12288 waves)
    {
        int waves  = 3 * H_ * ((B_ * S_) / 32);
        int blocks = waves / 4;
        mha_qkv_rope_kernel<<<blocks, 128, 0, stream>>>(xb, wqb, wkb, wvb, qb, kb, vtb);
    }

    // Stage 2: attention  (4*16*128 = 8192 waves)
    {
        int waves  = B_ * H_ * (S_ / 16);
        int blocks = waves / 4;
        mha_attn_kernel<<<blocks, 128, 0, stream>>>(qb, kb, vtb, ctxb);
    }

    // Stage 3: output projection (256 * 16 = 4096 waves)
    {
        int waves  = ((B_ * S_) / 32) * (D_ / 64);
        int blocks = waves / 4;
        mha_oproj_kernel<<<blocks, 128, 0, stream>>>(ctxb, wob, (float*)d_out);
    }
}
